// VN_DGCNN_santiago_27908697489758
// MI455X (gfx1250) — matrix operand order repaired and checked by reading, not compiled, not run on hardware
//
#include <hip/hip_runtime.h>

// ---------------------------------------------------------------------------
// VN-DGCNN forward for MI455X (gfx1250, wave32).
// Every matrix-shaped contraction (channel-mixing GEMMs + kNN Gram matrices)
// runs through one generic WMMA GEMM kernel (v_wmma_f32_16x16x32_f16, f32
// accumulate, f16 staged through LDS). The X tile is double-buffered with
// gfx1250 async global->LDS loads (ASYNCcnt) so DMA overlaps the WMMAs.
// Elementwise VN batchnorm / leaky-relu / pooling / gathers are VALU kernels
// (the net is HBM-bound at 23.3 TB/s).
// ---------------------------------------------------------------------------

typedef __attribute__((ext_vector_type(16))) _Float16 v16h;
typedef __attribute__((ext_vector_type(8)))  float    v8f;
typedef int vndg_b128 __attribute__((vector_size(16)));

#define GTM 64
#define GTN 64
#define GTK 32

#if defined(__gfx1250__) && __has_builtin(__builtin_amdgcn_global_load_async_to_lds_b128)
#define VNDG_ASYNC 1
#if __has_builtin(__builtin_amdgcn_s_wait_asynccnt)
#define VNDG_ASYNC_WAIT() __builtin_amdgcn_s_wait_asynccnt(0)
#else
#define VNDG_ASYNC_WAIT() asm volatile("s_wait_asynccnt 0" ::: "memory")
#endif
#else
#define VNDG_ASYNC 0
#define VNDG_ASYNC_WAIT() ((void)0)
#endif

// D[M,S] = A[M,K] (f32, row-major) * X[K,S] (f32, row-major), f16 WMMA, f32 acc.
// Double-buffered LDS pipeline: stage tile k+1 (async DMA for X) while the
// 2x2 wave grid runs its 4 WMMAs on tile k.
__global__ __launch_bounds__(128) void vndg_gemm(
    const float* __restrict__ A, const float* __restrict__ X,
    float* __restrict__ D, int M, int K, long S)
{
  __shared__ _Float16 As[2][GTM][GTK + 8];
#if VNDG_ASYNC
  __shared__ float Xs[2][GTK][GTN];        // async-filled f32 tiles
#else
  __shared__ _Float16 Xs[2][GTK][GTN + 8];
#endif
  const int  bm   = blockIdx.y * GTM;
  const long bn   = (long)blockIdx.x * GTN;
  const int  tid  = threadIdx.x;
  const int  wave = tid >> 5;
  const int  lane = tid & 31;
  const int  wm   = (wave >> 1) * 32;   // 2x2 waves -> 64x64 block tile
  const int  wn   = (wave & 1) * 32;
  const int  g    = lane >> 4;          // lane half
  const int  hl   = lane & 15;
  const bool fullM = (bm + GTM <= M);
  const bool fullN = (bn + GTN <= S);

  auto stage = [&](int buf, int k0) {
    const bool fullK = (k0 + GTK <= K);
    // ---- A tile (64x32 weights) f32 -> f16, LDS stores ----
    if (fullM && fullK) {
      for (int i = tid; i < GTM * GTK; i += 128) {
        int r = i >> 5, c = i & 31;
        As[buf][r][c] = (_Float16)A[(long)(bm + r) * K + (k0 + c)];
      }
    } else {
      for (int i = tid; i < GTM * GTK; i += 128) {
        int r = i >> 5, c = i & 31;
        int gm = bm + r, gk = k0 + c;
        As[buf][r][c] = (_Float16)((gm < M && gk < K) ? A[(long)gm * K + gk] : 0.0f);
      }
    }
    // ---- X tile (32x64) ----
#if VNDG_ASYNC
    if (fullK && fullN) {
      // 512 x b128 async global->LDS DMA transfers (ASYNCcnt tracked)
      for (int i = tid; i < (GTK * GTN) / 4; i += 128) {
        int r = i >> 4, c4 = (i & 15) * 4;
        const float* gp = X + (long)(k0 + r) * S + bn + c4;
        __builtin_amdgcn_global_load_async_to_lds_b128(
            (vndg_b128*)gp, (vndg_b128*)&Xs[buf][r][c4], 0, 0);
      }
    } else {
      for (int i = tid; i < GTK * GTN; i += 128) {
        int r = i >> 6, c = i & 63;
        int gk = k0 + r; long gs = bn + c;
        Xs[buf][r][c] = (gk < K && gs < S) ? X[(long)gk * S + gs] : 0.0f;
      }
    }
#else
    if (fullK && fullN) {
      // vectorized: global_load_b128 + packed cvt + 64-bit LDS stores
      for (int i = tid; i < (GTK * GTN) / 4; i += 128) {
        int r = i >> 4, c4 = (i & 15) * 4;
        const float4 v = *(const float4*)(X + (long)(k0 + r) * S + bn + c4);
        Xs[buf][r][c4 + 0] = (_Float16)v.x;
        Xs[buf][r][c4 + 1] = (_Float16)v.y;
        Xs[buf][r][c4 + 2] = (_Float16)v.z;
        Xs[buf][r][c4 + 3] = (_Float16)v.w;
      }
    } else {
      for (int i = tid; i < GTK * GTN; i += 128) {
        int r = i >> 6, c = i & 63;
        int gk = k0 + r; long gs = bn + c;
        Xs[buf][r][c] = (_Float16)((gk < K && gs < S) ? X[(long)gk * S + gs] : 0.0f);
      }
    }
#endif
  };

  v8f acc[2][2] = {};

  // Prologue: stage tile 0, wait for its async DMA, make it visible.
  stage(0, 0);
  VNDG_ASYNC_WAIT();
  __syncthreads();

  int buf = 0;
  for (int k0 = 0; k0 < K; k0 += GTK, buf ^= 1) {
    const int k1 = k0 + GTK;
    if (k1 < K) stage(buf ^ 1, k1);     // overlap next-tile DMA with WMMAs

    // Build fragments per CDNA5 ISA layouts (CDNA5 ISA §7.12.2).
    v16h af[2], bf[2];
#pragma unroll
    for (int t = 0; t < 2; ++t) {
      int mr = wm + t * 16 + hl;   // A: lane%16 = M row
      int nc = wn + t * 16 + hl;   // B: lane%16 = N col
#pragma unroll
      for (int v = 0; v < 8; ++v) {
        int ka = (v < 4 ? 2 * v : 16 + 2 * (v - 4)) + 8 * g;  // A K pattern
        af[t][2 * v]     = As[buf][mr][ka];
        af[t][2 * v + 1] = As[buf][mr][ka + 1];
        int kb = (v < 4 ? 2 * v : 16 + 2 * (v - 4)) + 8 * g;  // B K pattern
#if VNDG_ASYNC
        bf[t][2 * v]     = (_Float16)Xs[buf][kb][nc];
        bf[t][2 * v + 1] = (_Float16)Xs[buf][kb + 1][nc];
#else
        bf[t][2 * v]     = Xs[buf][kb][nc];
        bf[t][2 * v + 1] = Xs[buf][kb + 1][nc];
#endif
      }
    }
#pragma unroll
    for (int i = 0; i < 2; ++i)
#pragma unroll
      for (int j = 0; j < 2; ++j)
        acc[i][j] = __builtin_amdgcn_wmma_f32_16x16x32_f16(
            false, af[i], false, bf[j], (short)0, acc[i][j], false, false);

    if (k1 < K) {
      VNDG_ASYNC_WAIT();               // next tile's DMA landed in LDS
      __syncthreads();                 // visibility + safe buffer reuse
    }
  }

  // C/D layout: VGPR r -> M = r + 8*(lane/16), N = lane%16
#pragma unroll
  for (int i = 0; i < 2; ++i)
#pragma unroll
    for (int j = 0; j < 2; ++j) {
      long gn = bn + wn + j * 16 + hl;
      if (gn < S) {
#pragma unroll
        for (int r = 0; r < 8; ++r) {
          int gm = bm + wm + i * 16 + 8 * g + r;
          if (gm < M) D[(long)gm * S + gn] = acc[i][j][r];
        }
      }
    }
}

// ---------------------------------------------------------------------------
// kNN support: repack one batch into xt[N][D] / xp[D][N]; the Gram matrix is
// then a WMMA GEMM, and top-20 selection runs LDS-resident per query row.
// ---------------------------------------------------------------------------
__global__ void vndg_repack(const float* __restrict__ x, int C, int Bb, int N, int b,
                            float* __restrict__ xt, float* __restrict__ xp)
{
  const int Dd = C * 3;
  const long chs = (long)Bb * 3 * N;
  const long total = (long)N * Dd;
  for (long t = (long)blockIdx.x * blockDim.x + threadIdx.x; t < total;
       t += (long)gridDim.x * blockDim.x) {
    int m = (int)(t / Dd), d = (int)(t % Dd);
    int c = d / 3, j = d % 3;
    float v = x[(long)c * chs + ((long)b * 3 + j) * N + m];
    xt[(long)m * Dd + d] = v;
    xp[(long)d * N + m] = v;
  }
}

__global__ void vndg_xx(const float* __restrict__ xt, float* __restrict__ xx, int N, int Dd)
{
  int m = blockIdx.x * blockDim.x + threadIdx.x;
  if (m < N) {
    float s = 0.f;
    for (int d = 0; d < Dd; ++d) { float v = xt[(long)m * Dd + d]; s += v * v; }
    xx[m] = s;
  }
}

__global__ __launch_bounds__(256) void vndg_topk(
    const float* __restrict__ score, const float* __restrict__ xx,
    int N, int b, int Kk, int* __restrict__ idx)
{
  __shared__ float sc[4096];
  __shared__ float rv[256];
  __shared__ int   ri[256];
  const int n = blockIdx.x, tid = threadIdx.x;
  const float* row = score + (long)n * N;
  for (int m = tid; m < N; m += 256) sc[m] = 2.f * row[m] - xx[m];
  __syncthreads();
  for (int t = 0; t < Kk; ++t) {
    float best = -3.4e38f; int bi = 0;
    for (int m = tid; m < N; m += 256)
      if (sc[m] > best) { best = sc[m]; bi = m; }
    rv[tid] = best; ri[tid] = bi;
    __syncthreads();
    for (int s = 128; s > 0; s >>= 1) {
      if (tid < s && rv[tid + s] > rv[tid]) { rv[tid] = rv[tid + s]; ri[tid] = ri[tid + s]; }
      __syncthreads();
    }
    if (tid == 0) { idx[((long)b * N + n) * Kk + t] = ri[0]; sc[ri[0]] = -3.4e38f; }
    __syncthreads();
  }
}

// Edge features: out[2C][((b*3+j)*N+n)*Kk+kk]; ch<C: nbr-ctr, ch>=C: ctr.
__global__ void vndg_edge(const float* __restrict__ x, const int* __restrict__ idx,
                          float* __restrict__ out, int C, int Bb, int N, int Kk)
{
  const long chs_in  = (long)Bb * 3 * N;
  const long chs_out = (long)Bb * 3 * N * Kk;
  const long total   = 2L * C * chs_out;
  for (long t = (long)blockIdx.x * blockDim.x + threadIdx.x; t < total;
       t += (long)gridDim.x * blockDim.x) {
    int  ch = (int)(t / chs_out);
    long r  = t % chs_out;
    int  bj = (int)(r / ((long)N * Kk));
    long r2 = r % ((long)N * Kk);
    int  n  = (int)(r2 / Kk);
    int  kk = (int)(r2 % Kk);
    int  c  = ch % C;
    float cv = x[(long)c * chs_in + (long)bj * N + n];
    float v;
    if (ch >= C) v = cv;
    else {
      int b  = bj / 3;
      int nb = idx[((long)b * N + n) * Kk + kk];
      v = x[(long)c * chs_in + (long)bj * N + nb] - cv;
    }
    out[t] = v;
  }
}

// Per-channel sum / sumsq of vector norms (VNBatchNorm training stats).
__global__ __launch_bounds__(256) void vndg_vnstats(
    const float* __restrict__ p, float* __restrict__ sums, float* __restrict__ sumsq,
    int Bb, int COLS)
{
  const int c = blockIdx.y;
  const long chs = (long)Bb * 3 * COLS;
  const float* pc = p + (long)c * chs;
  const long per = (long)Bb * COLS;
  float ls = 0.f, lq = 0.f;
  for (long t = (long)blockIdx.x * 256 + threadIdx.x; t < per;
       t += (long)gridDim.x * 256) {
    int b = (int)(t / COLS), s = (int)(t % COLS);
    long base = ((long)b * 3) * COLS + s;
    float p0 = pc[base], p1 = pc[base + COLS], p2 = pc[base + 2L * COLS];
    float nm = sqrtf(p0 * p0 + p1 * p1 + p2 * p2) + 1e-6f;
    ls += nm; lq += nm * nm;
  }
  __shared__ float rs[256], rq[256];
  int tid = threadIdx.x;
  rs[tid] = ls; rq[tid] = lq;
  __syncthreads();
  for (int s = 128; s > 0; s >>= 1) {
    if (tid < s) { rs[tid] += rs[tid + s]; rq[tid] += rq[tid + s]; }
    __syncthreads();
  }
  if (tid == 0) { atomicAdd(&sums[c], rs[0]); atomicAdd(&sumsq[c], rq[0]); }
}

// VNBatchNorm + vector leaky-ReLU epilogue (p = Wf x, d = Wd x already done).
__global__ void vndg_vnapply(const float* __restrict__ p, const float* __restrict__ d,
                             float* __restrict__ out,
                             const float* __restrict__ sums, const float* __restrict__ sumsq,
                             const float* __restrict__ gamma, const float* __restrict__ beta,
                             int C, int Cd, int Bb, int COLS)
{
  const long per = (long)Bb * COLS;
  const long total = (long)C * per;
  const long chs = (long)Bb * 3 * COLS;
  const float cnt = (float)per;
  for (long t = (long)blockIdx.x * blockDim.x + threadIdx.x; t < total;
       t += (long)gridDim.x * blockDim.x) {
    int c = (int)(t / per);
    long r = t % per;
    int b = (int)(r / COLS), s = (int)(r % COLS);
    long base = ((long)b * 3) * COLS + s;
    const float* pc = p + (long)c * chs;
    const float* dc = d + (long)((Cd == 1) ? 0 : c) * chs;
    float p0 = pc[base], p1 = pc[base + COLS], p2 = pc[base + 2L * COLS];
    float nm = sqrtf(p0 * p0 + p1 * p1 + p2 * p2) + 1e-6f;
    float m  = sums[c] / cnt;
    float vv = sumsq[c] / cnt - m * m;
    float nbn = (nm - m) * rsqrtf(vv + 1e-5f) * gamma[c] + beta[c];
    float scl = nbn / nm;
    float q0 = p0 * scl, q1 = p1 * scl, q2 = p2 * scl;
    float d0 = dc[base], d1 = dc[base + COLS], d2 = dc[base + 2L * COLS];
    float dot = q0 * d0 + q1 * d1 + q2 * d2;
    float dsq = d0 * d0 + d1 * d1 + d2 * d2;
    float o0 = q0, o1 = q1, o2 = q2;
    if (dot < 0.f) {
      float tt = dot / (dsq + 1e-6f);
      o0 = q0 - tt * d0; o1 = q1 - tt * d1; o2 = q2 - tt * d2;
    }
    float* oc = out + (long)c * chs;
    oc[base]             = 0.2f * q0 + 0.8f * o0;
    oc[base + COLS]      = 0.2f * q1 + 0.8f * o1;
    oc[base + 2L * COLS] = 0.2f * q2 + 0.8f * o2;
  }
}

// VN max pool over neighbor axis: argmax_k <g, dp> (sum over 3 components).
__global__ void vndg_pool(const float* __restrict__ g, const float* __restrict__ dp,
                          float* __restrict__ xo, int C, int Bb, int N, int Kk)
{
  const long total = (long)C * Bb * N;
  const long chs = (long)Bb * 3 * N * Kk;
  const long js  = (long)N * Kk;
  const long chso = (long)Bb * 3 * N;
  for (long t = (long)blockIdx.x * blockDim.x + threadIdx.x; t < total;
       t += (long)gridDim.x * blockDim.x) {
    int c = (int)(t / ((long)Bb * N));
    long r = t % ((long)Bb * N);
    int b = (int)(r / N), n = (int)(r % N);
    const float* gc = g  + (long)c * chs;
    const float* dc = dp + (long)c * chs;
    long base = ((long)b * 3) * js + (long)n * Kk;
    float best = -3.4e38f; int bi = 0;
    for (int kk = 0; kk < Kk; ++kk) {
      float s = 0.f;
      for (int j = 0; j < 3; ++j)
        s += gc[base + j * js + kk] * dc[base + j * js + kk];
      if (s > best) { best = s; bi = kk; }
    }
    for (int j = 0; j < 3; ++j)
      xo[(long)c * chso + ((long)b * 3 + j) * N + n] = gc[base + j * js + bi];
  }
}

// Mean over N, broadcast into channels [Cin, 2*Cin) of the same buffer.
__global__ void vndg_mean(float* __restrict__ h2, int Cin, int Bb, int N)
{
  const int total = Cin * Bb * 3;
  const long chs = (long)Bb * 3 * N;
  for (int t = blockIdx.x * blockDim.x + threadIdx.x; t < total;
       t += gridDim.x * blockDim.x) {
    int c = t / (Bb * 3), bj = t % (Bb * 3);
    const float* src = h2 + (long)c * chs + (long)bj * N;
    float s = 0.f;
    for (int n = 0; n < N; ++n) s += src[n];
    s /= (float)N;
    float* dst = h2 + (long)(Cin + c) * chs + (long)bj * N;
    for (int n = 0; n < N; ++n) dst[n] = s;
  }
}

// Frame contraction: hz[i*3+k][b*N+n] = sum_j h2[i][(b3+j)N+n]*z0[j][(b3+k)N+n]
__global__ void vndg_frame(const float* __restrict__ h2, const float* __restrict__ z0,
                           float* __restrict__ hz, int Ci, int Bb, int N)
{
  const long cols = (long)Bb * N;
  const long total = (long)Ci * 3 * cols;
  const long chs = (long)Bb * 3 * N;
  for (long t = (long)blockIdx.x * blockDim.x + threadIdx.x; t < total;
       t += (long)gridDim.x * blockDim.x) {
    int ch = (int)(t / cols);
    long col = t % cols;
    int i = ch / 3, k = ch % 3;
    int b = (int)(col / N), n = (int)(col % N);
    float s = 0.f;
    for (int j = 0; j < 3; ++j)
      s += h2[(long)i * chs + ((long)b * 3 + j) * N + n] *
           z0[(long)j * chs + ((long)b * 3 + k) * N + n];
    hz[t] = s;
  }
}

// Scalar BatchNorm1d stats / apply (+ optional leaky-relu), layout [C][B*N].
__global__ __launch_bounds__(256) void vndg_cstats(
    const float* __restrict__ y, float* __restrict__ sums, float* __restrict__ sumsq, long cols)
{
  const int c = blockIdx.y;
  const float* yc = y + (long)c * cols;
  float ls = 0.f, lq = 0.f;
  for (long t = (long)blockIdx.x * 256 + threadIdx.x; t < cols;
       t += (long)gridDim.x * 256) {
    float v = yc[t]; ls += v; lq += v * v;
  }
  __shared__ float rs[256], rq[256];
  int tid = threadIdx.x;
  rs[tid] = ls; rq[tid] = lq;
  __syncthreads();
  for (int s = 128; s > 0; s >>= 1) {
    if (tid < s) { rs[tid] += rs[tid + s]; rq[tid] += rq[tid + s]; }
    __syncthreads();
  }
  if (tid == 0) { atomicAdd(&sums[c], rs[0]); atomicAdd(&sumsq[c], rq[0]); }
}

__global__ void vndg_capply(float* __restrict__ y,
                            const float* __restrict__ sums, const float* __restrict__ sumsq,
                            const float* __restrict__ gm, const float* __restrict__ bt,
                            int C, long cols, int relu)
{
  const long total = (long)C * cols;
  const float cnt = (float)cols;
  for (long t = (long)blockIdx.x * blockDim.x + threadIdx.x; t < total;
       t += (long)gridDim.x * blockDim.x) {
    int c = (int)(t / cols);
    float m = sums[c] / cnt;
    float vv = sumsq[c] / cnt - m * m;
    float v = (y[t] - m) * rsqrtf(vv + 1e-5f) * gm[c] + bt[c];
    if (relu && v < 0.f) v *= 0.2f;
    y[t] = v;
  }
}

// Row-wise L2-normalize classifier weight.
__global__ void vndg_wnorm(const float* __restrict__ w, float* __restrict__ wn, int R, int Cc)
{
  int r = blockIdx.x * blockDim.x + threadIdx.x;
  if (r < R) {
    float s = 0.f;
    for (int c = 0; c < Cc; ++c) { float v = w[(long)r * Cc + c]; s += v * v; }
    float inv = 1.0f / fmaxf(sqrtf(s), 1e-12f);
    for (int c = 0; c < Cc; ++c) wn[(long)r * Cc + c] = w[(long)r * Cc + c] * inv;
  }
}

// [C][B*N] channel-major -> (B,C,N) b-major output.
__global__ void vndg_trout(const float* __restrict__ in, float* __restrict__ out,
                           int C, int Bb, int N)
{
  const long cols = (long)Bb * N;
  const long total = (long)C * cols;
  for (long t = (long)blockIdx.x * blockDim.x + threadIdx.x; t < total;
       t += (long)gridDim.x * blockDim.x) {
    int c = (int)(t / cols);
    long col = t % cols;
    int b = (int)(col / N), n = (int)(col % N);
    out[((long)b * C + c) * N + n] = in[t];
  }
}

__global__ void vndg_zero(float* __restrict__ p, int n)
{
  int t = blockIdx.x * blockDim.x + threadIdx.x;
  if (t < n) p[t] = 0.f;
}

// ---------------------------------------------------------------------------

static inline unsigned nblk(long tot)
{
  long b = (tot + 255) / 256;
  if (b < 1) b = 1;
  if (b > (1L << 26)) b = (1L << 26);
  return (unsigned)b;
}

extern "C" void kernel_launch(void* const* d_in, const int* in_sizes, int n_in,
                              void* d_out, int out_size, void* d_ws, size_t ws_size,
                              hipStream_t stream)
{
  (void)in_sizes; (void)n_in; (void)out_size; (void)ws_size;

  const int  Bb = 4, N = 4096, KK = 20;
  const long SP = (long)Bb * 3 * N;   // 49152  cols per channel (per-point tensors)
  const long S1 = SP * KK;            // 983040 cols per channel (edge tensors)
  const long SC = (long)Bb * N;       // 16384  cols per channel (scalar tensors)

  auto W = [&](int i) { return (const float*)d_in[i]; };
  float* fw = (float*)d_ws;
  float* outp = (float*)d_out;

  // Workspace arena (float offsets). kNN score buffer and phase-2 buffers
  // recycle the phase-1 edge-feature slots (strictly sequential lifetimes).
  long o_idx   = 0;                     // B*N*KK ints
  long o_stats = o_idx + (long)Bb * N * KK;
  long o_wn    = o_stats + 1024;
  long o_x123  = o_wn + 13 * 256;
  long o_arena = o_x123 + 63 * SP;

  long oA = o_arena;                    // 42*S1 edge features
  long oB = oA + 42 * S1;               // 21*S1
  long oC = oB + 21 * S1;               // 21*S1
  long oD = oC + 21 * S1;               // 21*S1

  long o_score = o_arena;               // N*N Gram matrix (one batch at a time)
  long o_xt    = o_score + (long)N * N; // N*64
  long o_xp    = o_xt + (long)N * 64;   // 64*N
  long o_xx    = o_xp + (long)N * 64;   // N

  long oh2  = o_arena;                  // 682*SP
  long ot1  = oh2 + 682 * SP;           // 341*SP
  long ot2  = ot1 + 341 * SP;           // 341*SP
  long ozz  = ot2 + 341 * SP;           // 341*SP
  long oz0  = ozz + 341 * SP;           // 3*SP
  long ohz  = oz0 + 3 * SP;             // 2046*SC
  long ocb1 = ohz + 2046 * SC;          // 256*SC
  long ocb2 = ocb1 + 256 * SC;          // 256*SC
  long olg  = ocb2 + 256 * SC;          // 13*SC

  int*   idxp  = (int*)(fw + o_idx);
  float* statp = fw + o_stats;
  float* wnp   = fw + o_wn;
  float* x123p = fw + o_x123;
  float* fA = fw + oA;   float* fB = fw + oB;
  float* fC = fw + oC;   float* fD = fw + oD;
  float* scoreb = fw + o_score; float* xtb = fw + o_xt;
  float* xpb = fw + o_xp;       float* xxb = fw + o_xx;
  float* h2b  = fw + oh2;  float* t1b = fw + ot1; float* t2b = fw + ot2;
  float* zzb  = fw + ozz;  float* z0b = fw + oz0; float* hzb = fw + ohz;
  float* cb1  = fw + ocb1; float* cb2 = fw + ocb2; float* lgb = fw + olg;

  auto gemm = [&](const float* Aw, const float* Xx, float* Dd, int M, int Kd, long S) {
    dim3 g((unsigned)((S + GTN - 1) / GTN), (unsigned)((M + GTM - 1) / GTM));
    vndg_gemm<<<g, dim3(128), 0, stream>>>(Aw, Xx, Dd, M, Kd, S);
  };
  auto knn = [&](const float* x, int C) {
    int Dd = C * 3;
    for (int b = 0; b < Bb; ++b) {
      vndg_repack<<<nblk((long)N * Dd), 256, 0, stream>>>(x, C, Bb, N, b, xtb, xpb);
      vndg_xx<<<nblk(N), 256, 0, stream>>>(xtb, xxb, N, Dd);
      gemm(xtb, xpb, scoreb, N, Dd, N);   // WMMA Gram matrix
      vndg_topk<<<N, 256, 0, stream>>>(scoreb, xxb, N, b, KK, idxp);
    }
  };
  auto vnlayer = [&](float* p, float* dd, float* out, const float* gm, const float* bt,
                     int C, int Cd, int COLS) {
    vndg_zero<<<4, 256, 0, stream>>>(statp, 1024);
    long per = (long)Bb * COLS;
    unsigned gx = (unsigned)((per + 255) / 256); if (gx > 256) gx = 256;
    vndg_vnstats<<<dim3(gx, C), 256, 0, stream>>>(p, statp, statp + 512, Bb, COLS);
    vndg_vnapply<<<nblk((long)C * per), 256, 0, stream>>>(
        p, dd, out, statp, statp + 512, gm, bt, C, Cd, Bb, COLS);
  };
  auto convbn = [&](float* y, const float* gm, const float* bt, int C, int relu) {
    vndg_zero<<<4, 256, 0, stream>>>(statp, 1024);
    vndg_cstats<<<dim3(64, C), 256, 0, stream>>>(y, statp, statp + 512, SC);
    vndg_capply<<<nblk((long)C * SC), 256, 0, stream>>>(
        y, statp, statp + 512, gm, bt, C, SC, relu);
  };

  const float* X_IN = W(0);   // (B,3,N) == channel-major [1][B*3*N]

  // ---------------- Stage 1: edge conv on raw points ----------------
  knn(X_IN, 1);
  vndg_edge<<<nblk(2 * S1), 256, 0, stream>>>(X_IN, idxp, fA, 1, Bb, N, KK);
  gemm(W(1), fA, fB, 21, 2, S1);                 // c1 Wf
  gemm(W(2), fA, fC, 21, 2, S1);                 // c1 Wd
  vnlayer(fB, fC, fB, W(3), W(4), 21, 21, N * KK);
  gemm(W(5), fB, fC, 21, 21, S1);                // c2 Wf
  gemm(W(6), fB, fD, 21, 21, S1);                // c2 Wd
  vnlayer(fC, fD, fC, W(7), W(8), 21, 21, N * KK);
  gemm(W(25), fC, fD, 21, 21, S1);               // pool1 direction
  vndg_pool<<<nblk(21L * Bb * N), 256, 0, stream>>>(fC, fD, x123p, 21, Bb, N, KK);

  // ---------------- Stage 2 ----------------
  knn(x123p, 21);
  vndg_edge<<<nblk(42 * S1), 256, 0, stream>>>(x123p, idxp, fA, 21, Bb, N, KK);
  gemm(W(9),  fA, fB, 21, 42, S1);               // c3 Wf
  gemm(W(10), fA, fC, 21, 42, S1);               // c3 Wd
  vnlayer(fB, fC, fB, W(11), W(12), 21, 21, N * KK);
  gemm(W(13), fB, fC, 21, 21, S1);               // c4 Wf
  gemm(W(14), fB, fD, 21, 21, S1);               // c4 Wd
  vnlayer(fC, fD, fC, W(15), W(16), 21, 21, N * KK);
  gemm(W(26), fC, fD, 21, 21, S1);               // pool2 direction
  vndg_pool<<<nblk(21L * Bb * N), 256, 0, stream>>>(fC, fD, x123p + 21 * SP, 21, Bb, N, KK);

  // ---------------- Stage 3 ----------------
  knn(x123p + 21 * SP, 21);
  vndg_edge<<<nblk(42 * S1), 256, 0, stream>>>(x123p + 21 * SP, idxp, fA, 21, Bb, N, KK);
  gemm(W(17), fA, fB, 21, 42, S1);               // c5 Wf
  gemm(W(18), fA, fC, 21, 42, S1);               // c5 Wd
  vnlayer(fB, fC, fB, W(19), W(20), 21, 21, N * KK);
  gemm(W(27), fB, fC, 21, 21, S1);               // pool3 direction
  vndg_pool<<<nblk(21L * Bb * N), 256, 0, stream>>>(fB, fC, x123p + 42 * SP, 21, Bb, N, KK);

  // ---------------- Global head ----------------
  gemm(W(21), x123p, t1b, 341, 63, SP);          // c6 Wf
  gemm(W(22), x123p, t2b, 1, 63, SP);            // c6 Wd (shared)
  vnlayer(t1b, t2b, h2b, W(23), W(24), 341, 1, N);
  vndg_mean<<<nblk(341L * Bb * 3), 256, 0, stream>>>(h2b, 341, Bb, N);

  gemm(W(28), h2b, t1b, 341, 682, SP);           // s1 Wf
  gemm(W(29), h2b, t2b, 341, 682, SP);           // s1 Wd
  vnlayer(t1b, t2b, zzb, W(30), W(31), 341, 341, N);
  gemm(W(32), zzb, t1b, 170, 341, SP);           // s2 Wf
  gemm(W(33), zzb, t2b, 170, 341, SP);           // s2 Wd
  vnlayer(t1b, t2b, t1b, W(34), W(35), 170, 170, N);
  gemm(W(36), t1b, z0b, 3, 170, SP);             // s_lin -> frame z0

  vndg_frame<<<nblk(2046L * SC), 256, 0, stream>>>(h2b, z0b, hzb, 682, Bb, N);

  // ---------------- Conv head ----------------
  gemm(W(37), hzb, cb1, 256, 2046, SC); convbn(cb1, W(38), W(39), 256, 1);
  gemm(W(40), cb1, cb2, 256, 256,  SC); convbn(cb2, W(41), W(42), 256, 1);
  gemm(W(43), cb2, cb1, 128, 256,  SC); convbn(cb1, W(44), W(45), 128, 1);
  gemm(W(46), cb1, cb2, 256, 128,  SC); convbn(cb2, W(47), W(48), 256, 0); // features

  vndg_wnorm<<<1, 32, 0, stream>>>(W(49), wnp, 13, 256);
  gemm(wnp, cb2, lgb, 13, 256, SC);              // logits

  vndg_trout<<<nblk(13L * SC), 256, 0, stream>>>(lgb, outp, 13, Bb, N);
  vndg_trout<<<nblk(256L * SC), 256, 0, stream>>>(cb2, outp + 13L * SC, 256, Bb, N);
}
